// Lfm2DecoderLayer_32890859553430
// MI455X (gfx1250) — compile-verified, matrix operand order checked by reading, not run on hardware
//
#include <hip/hip_runtime.h>
#include <hip/hip_bf16.h>

// ---------------- problem constants ----------------
#define TT     2048
#define HIDD   2048
#define NH     32
#define NKV    8
#define HDIM   64
#define GRP    4            // NH / NKV
#define CACHED 1024
#define CTX    3072
#define INTER  8192
#define QKVN   3072         // NH*HD + 2*NKV*HD
#define BSLOT  16           // cache block size
#define ATTN_SCALE 0.125f   // HD^-0.5
#define RMS_EPS 1e-5f
#define LOG_THETA 13.815510558f  // ln(1e6)

typedef __bf16 bf16;
typedef __attribute__((ext_vector_type(16))) __bf16 bf16x16;
typedef __attribute__((ext_vector_type(8)))  float  floatx8;
typedef __attribute__((ext_vector_type(4)))  int    v4i;
typedef __attribute__((address_space(1))) v4i* gv4i_ptr;  // global int4*
typedef __attribute__((address_space(3))) v4i* lv4i_ptr;  // LDS int4*

#if __has_builtin(__builtin_amdgcn_global_load_async_to_lds_b128) && \
    __has_builtin(__builtin_amdgcn_s_wait_asynccnt)
#define USE_ASYNC_LDS 1
#else
#define USE_ASYNC_LDS 0
#endif

__device__ __forceinline__ floatx8 zero8() {
    floatx8 v;
#pragma unroll
    for (int i = 0; i < 8; ++i) v[i] = 0.0f;
    return v;
}

#if USE_ASYNC_LDS
__device__ __forceinline__ void async_copy16(const bf16* g, bf16* l) {
    __builtin_amdgcn_global_load_async_to_lds_b128(
        (gv4i_ptr)(v4i*)g, (lv4i_ptr)(v4i*)l, 0, 0);
}
#endif

// ---------------- fp32 -> bf16 convert ----------------
__global__ __launch_bounds__(256) void cvt_f32_bf16(const float* __restrict__ in,
                                                    bf16* __restrict__ out, int n) {
    int i = blockIdx.x * 256 + threadIdx.x;
    if (i < n) out[i] = (bf16)in[i];
}

// ---------------- row RMSNorm -> bf16 ----------------
__global__ __launch_bounds__(256) void rmsnorm_kernel(const float* __restrict__ x,
                                                      const float* __restrict__ w,
                                                      bf16* __restrict__ out, int ncols) {
    __shared__ float red[8];
    const int row = blockIdx.x;
    const float* xr = x + (size_t)row * ncols;
    float ss = 0.f;
    for (int i = threadIdx.x; i < ncols; i += 256) { float v = xr[i]; ss += v * v; }
#pragma unroll
    for (int d = 1; d < 32; d <<= 1) ss += __shfl_xor(ss, d, 32);
    if ((threadIdx.x & 31) == 0) red[threadIdx.x >> 5] = ss;
    __syncthreads();
    float tot = 0.f;
#pragma unroll
    for (int i = 0; i < 8; ++i) tot += red[i];
    const float rinv = rsqrtf(tot / (float)ncols + RMS_EPS);
    for (int i = threadIdx.x; i < ncols; i += 256)
        out[(size_t)row * ncols + i] = (bf16)(xr[i] * rinv * w[i]);
}

// ---------------- per-head RMS + RoPE for Q and new K ----------------
__global__ __launch_bounds__(32) void qk_rms_rope(const float* __restrict__ qkv,
                                                  const float* __restrict__ q_ln_w,
                                                  const float* __restrict__ k_ln_w,
                                                  const int* __restrict__ positions,
                                                  bf16* __restrict__ qb,
                                                  bf16* __restrict__ kfull) {
    const int t = blockIdx.x;
    const int head = blockIdx.y;
    const int lane = threadIdx.x;
    const bool isq = head < NH;
    const int kh = head - NH;
    const float* src = isq ? (qkv + (size_t)t * QKVN + head * HDIM)
                           : (qkv + (size_t)t * QKVN + NH * HDIM + kh * HDIM);
    const float* w = isq ? q_ln_w : k_ln_w;

    float x1 = src[lane];
    float x2 = src[lane + 32];
    float ss = x1 * x1 + x2 * x2;
#pragma unroll
    for (int d = 1; d < 32; d <<= 1) ss += __shfl_xor(ss, d, 32);
    const float rinv = rsqrtf(ss / (float)HDIM + RMS_EPS);
    x1 *= rinv * w[lane];
    x2 *= rinv * w[lane + 32];

    const float pos = (float)positions[t];
    const float inv = __expf(-((float)(2 * lane) / (float)HDIM) * LOG_THETA);
    float s, c;
    __sincosf(pos * inv, &s, &c);
    const float o1 = x1 * c - x2 * s;
    const float o2 = x2 * c + x1 * s;

    if (isq) {
        bf16* dst = qb + ((size_t)t * NH + head) * HDIM;
        dst[lane] = (bf16)o1;
        dst[lane + 32] = (bf16)o2;
    } else {
        bf16* dst = kfull + ((size_t)(CACHED + t) * NKV + kh) * HDIM;
        dst[lane] = (bf16)o1;
        dst[lane + 32] = (bf16)o2;
    }
}

// ---------------- gather cached K into k_full ----------------
__global__ __launch_bounds__(256) void build_k_cached(const float* __restrict__ kc,
                                                      const int* __restrict__ bt,
                                                      bf16* __restrict__ kf) {
    int i = blockIdx.x * 256 + threadIdx.x;
    if (i >= CACHED * NKV * HDIM) return;
    const int s = i / (NKV * HDIM);
    const int r = i % (NKV * HDIM);
    const int slot = bt[s / BSLOT] * BSLOT + (s % BSLOT);
    kf[i] = (bf16)kc[(size_t)slot * (NKV * HDIM) + r];
}

// ---------------- build transposed V: vt[kv*HD + d][ctx] ----------------
__global__ __launch_bounds__(256) void build_vT(const float* __restrict__ vc,
                                                const int* __restrict__ bt,
                                                const float* __restrict__ qkv,
                                                bf16* __restrict__ vt) {
    int i = blockIdx.x * 256 + threadIdx.x;
    if (i >= NKV * HDIM * CTX) return;
    const int s = i % CTX;        // context position (coalesced store)
    const int c = i / CTX;        // kv*HD + d
    float val;
    if (s < CACHED) {
        const int slot = bt[s / BSLOT] * BSLOT + (s % BSLOT);
        val = vc[(size_t)slot * (NKV * HDIM) + c];
    } else {
        val = qkv[(size_t)(s - CACHED) * QKVN + NH * HDIM + NKV * HDIM + c];
    }
    vt[i] = (bf16)val;
}

// ---------------- tiled bf16 WMMA GEMM: C[M,N] = A[M,K] * B[N,K]^T ----------------
// Block tile 128x128, BK=32, 8 waves each computing 64x32 (4x2 wmma tiles).
// Double-buffered LDS; async global->LDS when available.
// MODE 0: store f32. 1: store f32 + resid. 2: store bf16 silu(c). 3: store bf16 c*aux_bf16.
template <int MODE>
__global__ __launch_bounds__(256) void gemm_bf16_wmma(const bf16* __restrict__ A,
                                                      const bf16* __restrict__ B,
                                                      void* __restrict__ Cout,
                                                      const void* __restrict__ Aux,
                                                      int M, int N, int K) {
    __shared__ bf16 As[2][128 * 32];
    __shared__ bf16 Bs[2][128 * 32];
    const int tid = threadIdx.x;
    const int bm = blockIdx.y * 128;
    const int bn = blockIdx.x * 128;
    const int wave = tid >> 5;
    const int lane = tid & 31;
    const int m_base = (wave >> 2) * 64;   // 2 waves along M
    const int n_base = (wave & 3) * 32;    // 4 waves along N
    const int lrow = lane & 15;
    const int khalf = (lane >> 4) ? 8 : 0;  // K-offset for frags, row-offset for C

    // staging map: chunk -> (row, col16) within a 128x32 tile
    const int r0 = tid >> 2, c0 = (tid & 3) * 8;
    const int r1 = (tid + 256) >> 2, c1 = ((tid + 256) & 3) * 8;

    auto stage = [&](int buf, int k0) {
#if USE_ASYNC_LDS
        async_copy16(&A[(size_t)(bm + r0) * K + k0 + c0], &As[buf][r0 * 32 + c0]);
        async_copy16(&A[(size_t)(bm + r1) * K + k0 + c1], &As[buf][r1 * 32 + c1]);
        async_copy16(&B[(size_t)(bn + r0) * K + k0 + c0], &Bs[buf][r0 * 32 + c0]);
        async_copy16(&B[(size_t)(bn + r1) * K + k0 + c1], &Bs[buf][r1 * 32 + c1]);
#else
        *(uint4*)&As[buf][r0 * 32 + c0] = *(const uint4*)&A[(size_t)(bm + r0) * K + k0 + c0];
        *(uint4*)&As[buf][r1 * 32 + c1] = *(const uint4*)&A[(size_t)(bm + r1) * K + k0 + c1];
        *(uint4*)&Bs[buf][r0 * 32 + c0] = *(const uint4*)&B[(size_t)(bn + r0) * K + k0 + c0];
        *(uint4*)&Bs[buf][r1 * 32 + c1] = *(const uint4*)&B[(size_t)(bn + r1) * K + k0 + c1];
#endif
    };

    floatx8 acc[4][2];
#pragma unroll
    for (int i = 0; i < 4; ++i)
#pragma unroll
        for (int j = 0; j < 2; ++j) acc[i][j] = zero8();

    const int nIter = K >> 5;
    stage(0, 0);

    for (int it = 0; it < nIter; ++it) {
        const int cur = it & 1;
#if USE_ASYNC_LDS
        __builtin_amdgcn_s_wait_asynccnt(0);  // this wave's loads into `cur` done
#endif
        __syncthreads();  // everyone's `cur` ready; prior compute on `nxt` drained
        if (it + 1 < nIter) stage(cur ^ 1, (it + 1) << 5);

        bf16x16 aF[4], bF[2];
#pragma unroll
        for (int tm = 0; tm < 4; ++tm) {
            const bf16* p = &As[cur][(m_base + tm * 16 + lrow) * 32 + khalf];
#pragma unroll
            for (int i = 0; i < 8; ++i) { aF[tm][i] = p[i]; aF[tm][i + 8] = p[16 + i]; }
        }
#pragma unroll
        for (int tn = 0; tn < 2; ++tn) {
            const bf16* p = &Bs[cur][(n_base + tn * 16 + lrow) * 32 + khalf];
#pragma unroll
            for (int i = 0; i < 8; ++i) { bF[tn][i] = p[i]; bF[tn][i + 8] = p[16 + i]; }
        }
#pragma unroll
        for (int tm = 0; tm < 4; ++tm)
#pragma unroll
            for (int tn = 0; tn < 2; ++tn)
                acc[tm][tn] = __builtin_amdgcn_wmma_f32_16x16x32_bf16(
                    false, aF[tm], false, bF[tn], (short)0, acc[tm][tn], false, false);
    }

    // epilogue
#pragma unroll
    for (int tm = 0; tm < 4; ++tm)
#pragma unroll
        for (int tn = 0; tn < 2; ++tn)
#pragma unroll
            for (int r = 0; r < 8; ++r) {
                const int m = bm + m_base + tm * 16 + r + khalf;
                const int n = bn + n_base + tn * 16 + lrow;
                const size_t idx = (size_t)m * N + n;
                const float c = acc[tm][tn][r];
                if (MODE == 0) {
                    ((float*)Cout)[idx] = c;
                } else if (MODE == 1) {
                    ((float*)Cout)[idx] = c + ((const float*)Aux)[idx];
                } else if (MODE == 2) {
                    ((bf16*)Cout)[idx] = (bf16)(c / (1.0f + __expf(-c)));
                } else {
                    ((bf16*)Cout)[idx] = (bf16)(c * (float)((const bf16*)Aux)[idx]);
                }
            }
}

// ---------------- flash attention: 1 wave per (16-query block, head) ----------------
__global__ __launch_bounds__(32) void attn_kernel(const bf16* __restrict__ Qb,
                                                  const bf16* __restrict__ Kf,
                                                  const bf16* __restrict__ Vt,
                                                  bf16* __restrict__ Ob) {
    __shared__ bf16 Pl[16 * 32];
    const int t0 = blockIdx.x * 16;
    const int h = blockIdx.y;
    const int kv = h / GRP;
    const int lane = threadIdx.x;
    const int lrow = lane & 15;
    const int khalf = (lane >> 4) ? 8 : 0;
    const float NEG_INF = -__builtin_inff();

    bf16x16 qF[2];
#pragma unroll
    for (int j = 0; j < 2; ++j) {
        const bf16* p = Qb + ((size_t)(t0 + lrow) * NH + h) * HDIM + j * 32 + khalf;
#pragma unroll
        for (int i = 0; i < 8; ++i) { qF[j][i] = p[i]; qF[j][i + 8] = p[16 + i]; }
    }

    floatx8 o[4];
#pragma unroll
    for (int g = 0; g < 4; ++g) o[g] = zero8();
    float mrow[8], lsum[8];
#pragma unroll
    for (int r = 0; r < 8; ++r) { mrow[r] = NEG_INF; lsum[r] = 0.f; }

    const int rowt = t0 + khalf;            // query t for vgpr r = rowt + r
    const int kend = CACHED + t0 + 16;      // max key + 1

    for (int kb = 0; kb < kend; kb += 32) {
        float s[2][8];
#pragma unroll
        for (int u = 0; u < 2; ++u) {
            floatx8 sa = zero8();
#pragma unroll
            for (int j = 0; j < 2; ++j) {
                bf16x16 kF;
                const bf16* p =
                    Kf + ((size_t)(kb + u * 16 + lrow) * NKV + kv) * HDIM + j * 32 + khalf;
#pragma unroll
                for (int i = 0; i < 8; ++i) { kF[i] = p[i]; kF[i + 8] = p[16 + i]; }
                sa = __builtin_amdgcn_wmma_f32_16x16x32_bf16(false, qF[j], false, kF,
                                                             (short)0, sa, false, false);
            }
            const int key = kb + u * 16 + lrow;
#pragma unroll
            for (int r = 0; r < 8; ++r) {
                float v = sa[r] * ATTN_SCALE;
                if (key > CACHED + rowt + r) v = NEG_INF;  // causal mask
                s[u][r] = v;
            }
        }

        // online softmax (row stats live within each 16-lane half)
#pragma unroll
        for (int r = 0; r < 8; ++r) {
            float mx = fmaxf(s[0][r], s[1][r]);
#pragma unroll
            for (int d = 1; d < 16; d <<= 1) mx = fmaxf(mx, __shfl_xor(mx, d, 32));
            const float newm = fmaxf(mrow[r], mx);
            const float alpha = __expf(mrow[r] - newm);
            mrow[r] = newm;
            const float p0 = __expf(s[0][r] - newm);
            const float p1 = __expf(s[1][r] - newm);
            float ps = p0 + p1;
#pragma unroll
            for (int d = 1; d < 16; d <<= 1) ps += __shfl_xor(ps, d, 32);
            lsum[r] = lsum[r] * alpha + ps;
#pragma unroll
            for (int g = 0; g < 4; ++g) o[g][r] *= alpha;
            Pl[(r + khalf) * 32 + lrow] = (bf16)p0;
            Pl[(r + khalf) * 32 + 16 + lrow] = (bf16)p1;
        }
        __syncthreads();

        bf16x16 pF;
        {
            const bf16* pp = &Pl[lrow * 32 + khalf];
#pragma unroll
            for (int i = 0; i < 8; ++i) { pF[i] = pp[i]; pF[i + 8] = pp[16 + i]; }
        }
#pragma unroll
        for (int g = 0; g < 4; ++g) {
            bf16x16 vF;
            const bf16* vp = Vt + ((size_t)(kv * HDIM + g * 16 + lrow)) * CTX + kb + khalf;
#pragma unroll
            for (int i = 0; i < 8; ++i) { vF[i] = vp[i]; vF[i + 8] = vp[16 + i]; }
            o[g] = __builtin_amdgcn_wmma_f32_16x16x32_bf16(false, pF, false, vF, (short)0,
                                                           o[g], false, false);
        }
        __syncthreads();
    }

#pragma unroll
    for (int r = 0; r < 8; ++r) {
        const float inv = 1.0f / lsum[r];
#pragma unroll
        for (int g = 0; g < 4; ++g)
            Ob[(size_t)(rowt + r) * (NH * HDIM) + h * HDIM + g * 16 + lrow] =
                (bf16)(o[g][r] * inv);
    }
}

// ---------------- host side ----------------
extern "C" void kernel_launch(void* const* d_in, const int* in_sizes, int n_in,
                              void* d_out, int out_size, void* d_ws, size_t ws_size,
                              hipStream_t stream) {
    const float* hidden = (const float*)d_in[0];
    const float* k_cache = (const float*)d_in[1];
    const float* v_cache = (const float*)d_in[2];
    const float* op_norm_w = (const float*)d_in[3];
    const float* qkv_w = (const float*)d_in[4];
    const float* q_ln_w = (const float*)d_in[5];
    const float* k_ln_w = (const float*)d_in[6];
    const float* out_proj_w = (const float*)d_in[7];
    const float* ffn_norm_w = (const float*)d_in[8];
    const float* w1_w = (const float*)d_in[9];
    const float* w3_w = (const float*)d_in[10];
    const float* w2_w = (const float*)d_in[11];
    const int* positions = (const int*)d_in[12];
    const int* block_table = (const int*)d_in[14];
    float* out = (float*)d_out;

    size_t off = 0;
    auto carve = [&](size_t bytes) {
        void* p = (char*)d_ws + off;
        off += (bytes + 255) & ~(size_t)255;
        return p;
    };
    bf16* qkvw_bf = (bf16*)carve((size_t)QKVN * HIDD * 2);
    bf16* outw_bf = (bf16*)carve((size_t)HIDD * HIDD * 2);
    bf16* w1_bf = (bf16*)carve((size_t)INTER * HIDD * 2);
    bf16* w3_bf = (bf16*)carve((size_t)INTER * HIDD * 2);
    bf16* w2_bf = (bf16*)carve((size_t)HIDD * INTER * 2);
    bf16* h_bf = (bf16*)carve((size_t)TT * HIDD * 2);
    bf16* h2_bf = (bf16*)carve((size_t)TT * HIDD * 2);
    bf16* q_bf = (bf16*)carve((size_t)TT * NH * HDIM * 2);
    bf16* attn_bf = (bf16*)carve((size_t)TT * NH * HDIM * 2);
    bf16* silu_bf = (bf16*)carve((size_t)TT * INTER * 2);
    bf16* act_bf = (bf16*)carve((size_t)TT * INTER * 2);
    bf16* k_full = (bf16*)carve((size_t)CTX * NKV * HDIM * 2);
    bf16* vT = (bf16*)carve((size_t)NKV * HDIM * CTX * 2);
    float* qkv_f32 = (float*)carve((size_t)TT * QKVN * 4);
    float* x_f32 = (float*)carve((size_t)TT * HIDD * 4);
    (void)ws_size; (void)n_in; (void)in_sizes; (void)out_size;

    auto cblk = [](size_t n) { return (unsigned)((n + 255) / 256); };

    // weights -> bf16
    cvt_f32_bf16<<<cblk((size_t)QKVN * HIDD), 256, 0, stream>>>(qkv_w, qkvw_bf, QKVN * HIDD);
    cvt_f32_bf16<<<cblk((size_t)HIDD * HIDD), 256, 0, stream>>>(out_proj_w, outw_bf, HIDD * HIDD);
    cvt_f32_bf16<<<cblk((size_t)INTER * HIDD), 256, 0, stream>>>(w1_w, w1_bf, INTER * HIDD);
    cvt_f32_bf16<<<cblk((size_t)INTER * HIDD), 256, 0, stream>>>(w3_w, w3_bf, INTER * HIDD);
    cvt_f32_bf16<<<cblk((size_t)HIDD * INTER), 256, 0, stream>>>(w2_w, w2_bf, HIDD * INTER);

    // pre-attn norm
    rmsnorm_kernel<<<TT, 256, 0, stream>>>(hidden, op_norm_w, h_bf, HIDD);

    // QKV projection
    gemm_bf16_wmma<0><<<dim3(QKVN / 128, TT / 128), 256, 0, stream>>>(
        h_bf, qkvw_bf, (void*)qkv_f32, nullptr, TT, QKVN, HIDD);

    // per-head RMS + RoPE (Q, new K)
    qk_rms_rope<<<dim3(TT, NH + NKV), 32, 0, stream>>>(qkv_f32, q_ln_w, k_ln_w, positions,
                                                       q_bf, k_full);
    // cached K gather + transposed V build
    build_k_cached<<<cblk((size_t)CACHED * NKV * HDIM), 256, 0, stream>>>(k_cache,
                                                                          block_table, k_full);
    build_vT<<<cblk((size_t)NKV * HDIM * CTX), 256, 0, stream>>>(v_cache, block_table,
                                                                 qkv_f32, vT);

    // attention
    attn_kernel<<<dim3(TT / 16, NH), 32, 0, stream>>>(q_bf, k_full, vT, attn_bf);

    // out projection + residual
    gemm_bf16_wmma<1><<<dim3(HIDD / 128, TT / 128), 256, 0, stream>>>(
        attn_bf, outw_bf, (void*)x_f32, (const void*)hidden, TT, HIDD, HIDD);

    // FFN
    rmsnorm_kernel<<<TT, 256, 0, stream>>>(x_f32, ffn_norm_w, h2_bf, HIDD);
    gemm_bf16_wmma<2><<<dim3(INTER / 128, TT / 128), 256, 0, stream>>>(
        h2_bf, w1_bf, (void*)silu_bf, nullptr, TT, INTER, HIDD);
    gemm_bf16_wmma<3><<<dim3(INTER / 128, TT / 128), 256, 0, stream>>>(
        h2_bf, w3_bf, (void*)act_bf, (const void*)silu_bf, TT, INTER, HIDD);
    gemm_bf16_wmma<1><<<dim3(HIDD / 128, TT / 128), 256, 0, stream>>>(
        act_bf, w2_bf, (void*)out, (const void*)x_f32, TT, HIDD, INTER);
}
